// LSTM_67748814127482
// MI455X (gfx1250) — matrix operand order repaired and checked by reading, not compiled, not run on hardware
//
#include <hip/hip_runtime.h>
#include <hip/hip_bf16.h>

// ---------------------------------------------------------------------------
// 2-layer LSTM (B=16, T=1024, H=1024, SIZE=256, OUT=256) for gfx1250.
// Strategy:
//  * Pre-pack W0/W1/Wd to BF16 WMMA-fragment layout; convert X to BF16.
//  * Persistent kernel: 64 blocks (one 16-wide hidden tile each) x 128 thr
//    (4 waves, K split 4-way, LDS reduction). c & fp32 h live in LDS for the
//    whole sequence. h broadcast between blocks via global BF16 buffers,
//    double-buffered, with a global atomic grid barrier (2 per step).
//  * Batch row 15's h1 logged per step -> final WMMA GEMM for logits.
// ---------------------------------------------------------------------------

typedef __attribute__((ext_vector_type(16))) __bf16          v16bf;
typedef __attribute__((ext_vector_type(8)))  float           v8f;
typedef __attribute__((ext_vector_type(8)))  unsigned short  us8;

#define NBLOCKS 64          // one per hidden tile (H/16)

__device__ __forceinline__ unsigned short f2bf(float f) {
  unsigned u = __builtin_bit_cast(unsigned, f);
  u += 0x7FFFu + ((u >> 16) & 1u);          // round-to-nearest-even
  return (unsigned short)(u >> 16);
}

__device__ __forceinline__ float sigm(float x) { return 1.0f / (1.0f + __expf(-x)); }

// A-matrix fragment (16x32 bf16) from row-major bf16 [16][ldk].
// ISA layout: lane L: m = L&15, half = L>>4; VGPR0-3 = K k0+8h..+7,
// VGPR4-7 = K k0+16+8h..+7  => two contiguous 16B chunks.
__device__ __forceinline__ v16bf load_a_frag(const unsigned short* base, int ldk,
                                             int k0, int lane) {
  const int m = lane & 15, h = lane >> 4;
  const unsigned short* p = base + (size_t)m * ldk + (k0 + 8 * h);
  union { us8 u[2]; v16bf v; } x;
  x.u[0] = *(const us8*)(p);
  x.u[1] = *(const us8*)(p + 16);
  return x.v;
}

// B-matrix fragment (32x16 bf16) from pre-packed tile: [tile][lane][16] bf16.
__device__ __forceinline__ v16bf load_b_frag(const unsigned short* tile, int lane) {
  const unsigned short* p = tile + lane * 16;
  union { us8 u[2]; v16bf v; } x;
  x.u[0] = *(const us8*)(p);
  x.u[1] = *(const us8*)(p + 8);
  return x.v;
}

// Global software barrier (all NBLOCKS blocks resident: 64 blocks << #WGPs).
__device__ __forceinline__ void grid_barrier(unsigned* cnt, unsigned* gen) {
  __threadfence();                 // flush this wave's global stores to agent scope
  __syncthreads();
  if (threadIdx.x == 0) {
    unsigned g = __hip_atomic_load(gen, __ATOMIC_RELAXED, __HIP_MEMORY_SCOPE_AGENT);
    unsigned a = __hip_atomic_fetch_add(cnt, 1u, __ATOMIC_ACQ_REL, __HIP_MEMORY_SCOPE_AGENT);
    if (a == NBLOCKS - 1u) {
      __hip_atomic_store(cnt, 0u, __ATOMIC_RELAXED, __HIP_MEMORY_SCOPE_AGENT);
      __hip_atomic_fetch_add(gen, 1u, __ATOMIC_RELEASE, __HIP_MEMORY_SCOPE_AGENT);
    } else {
      while (__hip_atomic_load(gen, __ATOMIC_ACQUIRE, __HIP_MEMORY_SCOPE_AGENT) == g)
        __builtin_amdgcn_s_sleep(1);
    }
  }
  __syncthreads();
}

// One LSTM layer step for this block's 16 hidden units.
// A = [seg0 | seg1] along K; waves split K 4-way; 4 gate accumulators each.
__device__ __forceinline__ void layer_phase(
    int wave, int lane, int ht,
    const unsigned short* a0, int a0ldk, int a0len,   // first A segment
    const unsigned short* a1, int a1ldk,              // second A segment
    const unsigned short* Wf, int KT,                 // packed weights, K/32
    const float* bias,
    float (*red)[4][32][8],                           // [wave][gate][lane][r]
    float (*ct)[8], float (*hti)[8],                  // LDS c / h tiles
    unsigned short* hdst,                             // bf16 [16][1024] h out
    unsigned short* ydst)                             // bf16 Y row (or null)
{
  const int ktiles = KT >> 2;
  const int kbeg   = wave * ktiles;
  v8f acc[4] = { {}, {}, {}, {} };

  for (int kt = kbeg; kt < kbeg + ktiles; ++kt) {
    const int k0 = kt << 5;
    v16bf a = (k0 < a0len) ? load_a_frag(a0, a0ldk, k0, lane)
                           : load_a_frag(a1, a1ldk, k0 - a0len, lane);
#pragma unroll
    for (int g = 0; g < 4; ++g) {
      const unsigned short* wt = Wf + ((size_t)((g * 64 + ht) * KT + kt) << 9);
      v16bf b = load_b_frag(wt, lane);
      acc[g] = __builtin_amdgcn_wmma_f32_16x16x32_bf16(
          false, a, false, b, (short)0, acc[g], false, false);
    }
  }
#pragma unroll
  for (int g = 0; g < 4; ++g)
#pragma unroll
    for (int r = 0; r < 8; ++r)
      red[wave][g][lane][r] = acc[g][r];
  __syncthreads();

  if (wave == 0) {
    const int n = lane & 15, half = lane >> 4;
    const int u = ht * 16 + n;
    float gv[4][8];
#pragma unroll
    for (int g = 0; g < 4; ++g) {
      const float bg = bias[g * 1024 + u];
#pragma unroll
      for (int r = 0; r < 8; ++r)
        gv[g][r] = red[0][g][lane][r] + red[1][g][lane][r] +
                   red[2][g][lane][r] + red[3][g][lane][r] + bg;
    }
#pragma unroll
    for (int r = 0; r < 8; ++r) {
      // gate order i, j, f, o (TF LSTMCell); forget bias = 1.0
      const float c  = ct[lane][r];
      const float nc = sigm(gv[2][r] + 1.0f) * c + sigm(gv[0][r]) * tanhf(gv[1][r]);
      const float nh = sigm(gv[3][r]) * tanhf(nc);
      ct[lane][r]  = nc;
      hti[lane][r] = nh;
      const int m = r + half * 8;
      const unsigned short hb = f2bf(nh);
      hdst[m * 1024 + u] = hb;
      if (ydst && m == 15) ydst[u] = hb;      // batch row 15 -> logits input
    }
  }
  // grid_barrier follows from the caller (contains __syncthreads)
}

__global__ __launch_bounds__(128) void lstm_persist(
    const unsigned short* __restrict__ Xbf,      // [16][1024][256] bf16
    const float* __restrict__ state,             // [16][4096] f32
    const unsigned short* __restrict__ W0f, const float* __restrict__ b0,
    const unsigned short* __restrict__ W1f, const float* __restrict__ b1,
    unsigned short* __restrict__ h0bf,           // [2][16][1024] bf16
    unsigned short* __restrict__ h1bf,           // [2][16][1024] bf16
    unsigned short* __restrict__ Ybf,            // [1024][1024] bf16
    float* __restrict__ out_state,               // [16][4096] f32
    unsigned* __restrict__ bar)
{
  __shared__ float red[4][4][32][8];             // 16 KB K-split partials
  __shared__ float c0t[32][8], h0t[32][8], c1t[32][8], h1t[32][8];

  const int tid  = threadIdx.x;
  const int wave = tid >> 5, lane = tid & 31;
  const int ht   = blockIdx.x;                   // hidden tile 0..63
  const int nloc = lane & 15, half = lane >> 4;
  unsigned* cnt = bar;
  unsigned* gen = bar + 1;

  // ---- init: load state = [c0 | h0 | c1 | h1] for this tile ----
  {
    const int u = ht * 16 + nloc;
#pragma unroll
    for (int r = 0; r < 8; ++r) {
      const int m = r + half * 8;
      const float* srow = state + (size_t)m * 4096;
      if (wave == 0) c0t[lane][r] = srow[u];
      if (wave == 1) { float v = srow[1024 + u]; h0t[lane][r] = v;
                       h0bf[16384 + m * 1024 + u] = f2bf(v); }
      if (wave == 2) c1t[lane][r] = srow[2048 + u];
      if (wave == 3) { float v = srow[3072 + u]; h1t[lane][r] = v;
                       h1bf[16384 + m * 1024 + u] = f2bf(v); }
    }
  }
  grid_barrier(cnt, gen);

  for (int t = 0; t < 1024; ++t) {
    const int p = t & 1, q = p ^ 1;
    // layer 0: gates = [x_t, h0(t-1)] @ W0 + b0     (K = 256 + 1024)
    layer_phase(wave, lane, ht,
                Xbf + (size_t)t * 256, 262144 /*T*SIZE*/, 256,
                h0bf + q * 16384, 1024,
                W0f, 40, b0, red, c0t, h0t,
                h0bf + p * 16384, nullptr);
    grid_barrier(cnt, gen);
    // layer 1: gates = [h0(t), h1(t-1)] @ W1 + b1   (K = 1024 + 1024)
    layer_phase(wave, lane, ht,
                h0bf + p * 16384, 1024, 1024,
                h1bf + q * 16384, 1024,
                W1f, 64, b1, red, c1t, h1t,
                h1bf + p * 16384, Ybf + (size_t)t * 1024);
    grid_barrier(cnt, gen);
  }

  // ---- final state: concat([c0, h0, c1, h1]) ----
  {
    const int u = ht * 16 + nloc;
    const float (*tp)[8] = (wave == 0) ? c0t : (wave == 1) ? h0t
                          : (wave == 2) ? c1t : h1t;
#pragma unroll
    for (int r = 0; r < 8; ++r) {
      const int m = r + half * 8;
      out_state[(size_t)m * 4096 + wave * 1024 + u] = tp[lane][r];
    }
  }
}

// logits[1024][256] = Ybf[1024][1024] @ Wd + bd, via WMMA (64x16 tiles).
__global__ __launch_bounds__(128) void dense_out_k(
    const unsigned short* __restrict__ Ybf,
    const unsigned short* __restrict__ Wdf,
    const float* __restrict__ bd,
    float* __restrict__ logits)
{
  const int lane = threadIdx.x & 31;
  const int gw = blockIdx.x * (blockDim.x >> 5) + (threadIdx.x >> 5);
  const int nw = gridDim.x * (blockDim.x >> 5);
  for (int tile = gw; tile < 64 * 16; tile += nw) {
    const int mt = tile >> 4, nt = tile & 15;
    v8f acc = {};
    const unsigned short* arow = Ybf + (size_t)mt * 16 * 1024;
    for (int kt = 0; kt < 32; ++kt) {
      v16bf a = load_a_frag(arow, 1024, kt << 5, lane);
      v16bf b = load_b_frag(Wdf + ((size_t)(nt * 32 + kt) << 9), lane);
      acc = __builtin_amdgcn_wmma_f32_16x16x32_bf16(
          false, a, false, b, (short)0, acc, false, false);
    }
    const int n = lane & 15, half = lane >> 4;
    const float bg = bd[nt * 16 + n];
#pragma unroll
    for (int r = 0; r < 8; ++r) {
      const int m = r + half * 8;
      logits[(size_t)(mt * 16 + m) * 256 + nt * 16 + n] = acc[r] + bg;
    }
  }
}

// Pack row-major f32 W[K][N] into bf16 WMMA-B fragment layout:
// tile (nt,kt) holds 32 lanes x 16 bf16 contiguous; lane L -> col nt*16+(L&15),
// K = kt*32 + 16*(j>>3) + 8*(L>>4) + (j&7).
__global__ void pack_w_bf16(const float* __restrict__ W,
                            unsigned short* __restrict__ out, int K, int N) {
  const int KT = K >> 5;
  const size_t total = (size_t)K * N;
  for (size_t idx = (size_t)blockIdx.x * blockDim.x + threadIdx.x; idx < total;
       idx += (size_t)gridDim.x * blockDim.x) {
    const size_t tile = idx >> 9;
    const int rem = (int)(idx & 511);
    const int lane = rem >> 4, j = rem & 15;
    const int nt = (int)(tile / KT), kt = (int)(tile % KT);
    const int n = nt * 16 + (lane & 15);
    const int k = (kt << 5) + ((j >> 3) << 4) + ((lane >> 4) << 3) + (j & 7);
    out[idx] = f2bf(W[(size_t)k * N + n]);
  }
}

__global__ void conv_bf16(const float* __restrict__ in,
                          unsigned short* __restrict__ out, size_t n) {
  for (size_t i = (size_t)blockIdx.x * blockDim.x + threadIdx.x; i < n;
       i += (size_t)gridDim.x * blockDim.x)
    out[i] = f2bf(in[i]);
}

extern "C" void kernel_launch(void* const* d_in, const int* in_sizes, int n_in,
                              void* d_out, int out_size, void* d_ws, size_t ws_size,
                              hipStream_t stream) {
  (void)in_sizes; (void)n_in; (void)out_size; (void)ws_size;
  const float* X  = (const float*)d_in[0];
  const float* st = (const float*)d_in[1];
  const float* W0 = (const float*)d_in[2];
  const float* b0 = (const float*)d_in[3];
  const float* W1 = (const float*)d_in[4];
  const float* b1 = (const float*)d_in[5];
  const float* Wd = (const float*)d_in[6];
  const float* bd = (const float*)d_in[7];

  // workspace carve (bytes, 256-aligned); total ~38.4 MB
  char* ws = (char*)d_ws;
  unsigned*       bar  = (unsigned*)      (ws + 0);
  unsigned short* W0f  = (unsigned short*)(ws + 256);                 // 10,485,760
  unsigned short* W1f  = (unsigned short*)(ws + 10486016);            // 16,777,216
  unsigned short* Wdf  = (unsigned short*)(ws + 27263232);            //    524,288
  unsigned short* Xbf  = (unsigned short*)(ws + 27787520);            //  8,388,608
  unsigned short* h0bf = (unsigned short*)(ws + 36176128);            //     65,536
  unsigned short* h1bf = (unsigned short*)(ws + 36241664);            //     65,536
  unsigned short* Ybf  = (unsigned short*)(ws + 36307200);            //  2,097,152

  hipMemsetAsync(bar, 0, 256, stream);
  pack_w_bf16<<<4096, 256, 0, stream>>>(W0, W0f, 1280, 4096);
  pack_w_bf16<<<4096, 256, 0, stream>>>(W1, W1f, 2048, 4096);
  pack_w_bf16<<<512,  256, 0, stream>>>(Wd, Wdf, 1024, 256);
  conv_bf16<<<4096, 256, 0, stream>>>(X, Xbf, (size_t)16 * 1024 * 256);

  float* logits    = (float*)d_out;            // [1024][256]
  float* out_state = logits + 1024 * 256;      // [16][4096]
  lstm_persist<<<NBLOCKS, 128, 0, stream>>>(Xbf, st, W0f, b0, W1f, b1,
                                            h0bf, h1bf, Ybf, out_state, bar);
  dense_out_k<<<128, 128, 0, stream>>>(Ybf, Wdf, bd, logits);
}
